// DecoderRNN_21869973471764
// MI455X (gfx1250) — compile-verified, matrix operand order checked by reading, not run on hardware
//
#include <hip/hip_runtime.h>

#define EMBED  512
#define HIDDEN 512
#define VOCAB  10000
#define BATCH  64
#define SEQ    32
#define G4     2048   // 4*HIDDEN
#define MROWS  2048   // SEQ*BATCH

typedef __bf16 bf16;
typedef __attribute__((ext_vector_type(16))) bf16  v16bf;
typedef __attribute__((ext_vector_type(8)))  bf16  v8bf;
typedef __attribute__((ext_vector_type(8)))  float v8f;
typedef __attribute__((ext_vector_type(4)))  unsigned int u32x4;
typedef __attribute__((ext_vector_type(8)))  unsigned int u32x8;

// Load one 16x32 bf16 WMMA fragment row-slice for this lane.
// rowk points at element (row, kbase) of a K-contiguous row.  ISA layout:
// halves 0..7  = K: hi*8 + 0..7       (VGPRs 0-3)
// halves 8..15 = K: 16 + hi*8 + 0..7  (VGPRs 4-7)
__device__ inline v16bf load_frag(const bf16* rowk, int hi) {
    v8bf lo = *(const v8bf*)(rowk + hi * 8);
    v8bf hh = *(const v8bf*)(rowk + 16 + hi * 8);
    return __builtin_shufflevector(lo, hh, 0, 1, 2, 3, 4, 5, 6, 7,
                                   8, 9, 10, 11, 12, 13, 14, 15);
}

__device__ inline float sigmoidf(float x) { return 1.0f / (1.0f + __expf(-x)); }

// ---------------------------------------------------------------- converters
__global__ void cvt_bf16_kernel(const float* __restrict__ src,
                                bf16* __restrict__ dst, int n) {
    int i = blockIdx.x * blockDim.x + threadIdx.x;
    int stride = gridDim.x * blockDim.x;
    for (; i < n; i += stride) dst[i] = (bf16)src[i];
}

// Build xs rows (r = t*64+b): t==0 -> features[b], else emb[captions[b][t-1]]
__global__ void build_xs_kernel(const float* __restrict__ feat,
                                const int* __restrict__ caps,
                                const float* __restrict__ emb,
                                bf16* __restrict__ xs) {
    int idx = blockIdx.x * blockDim.x + threadIdx.x;
    int stride = gridDim.x * blockDim.x;
    const int total = MROWS * EMBED;
    for (; idx < total; idx += stride) {
        int r = idx >> 9;            // row (t*64+b)
        int e = idx & 511;
        int t = r >> 6, b = r & 63;
        float v;
        if (t == 0) {
            v = feat[b * EMBED + e];
        } else {
            int tok = caps[b * SEQ + (t - 1)];
            v = emb[tok * EMBED + e];
        }
        xs[idx] = (bf16)v;
    }
}

// ------------------------------------------------- x_gates = xs @ W_ih^T + b
// M=2048, N=2048, K=512.  One wave computes a 16x64 tile (A reused 4x).
// K-loop is software-pipelined; unroll-by-2 lets the two fragment buffers
// alternate roles so the rotation copies vanish.
__global__ __launch_bounds__(256) void xgates_gemm_kernel(
    const bf16* __restrict__ xs, const bf16* __restrict__ wih,
    const float* __restrict__ bih, const float* __restrict__ bhh,
    float* __restrict__ xg) {
    int wave = (blockIdx.x * blockDim.x + threadIdx.x) >> 5;
    int lane = threadIdx.x & 31;
    int mt = wave >> 5;          // 0..127
    int ng = wave & 31;          // 0..31 (groups of 4 N-tiles)
    int mb = mt * 16, nb = ng * 64;
    int l15 = lane & 15, hi = lane >> 4;

    v8f acc[4] = {};
    const bf16* arow = xs + (size_t)(mb + l15) * EMBED;
    const bf16* brow[4];
#pragma unroll
    for (int j = 0; j < 4; ++j)
        brow[j] = wih + (size_t)(nb + j * 16 + l15) * EMBED;

    v16bf a = load_frag(arow, hi);
    v16bf b[4];
#pragma unroll
    for (int j = 0; j < 4; ++j) b[j] = load_frag(brow[j], hi);

#pragma unroll 2
    for (int k = 32; k < EMBED; k += 32) {
        v16bf an = load_frag(arow + k, hi);
        v16bf bn[4];
#pragma unroll
        for (int j = 0; j < 4; ++j) bn[j] = load_frag(brow[j] + k, hi);
#pragma unroll
        for (int j = 0; j < 4; ++j)
            acc[j] = __builtin_amdgcn_wmma_f32_16x16x32_bf16(
                false, a, false, b[j], (short)0, acc[j], false, false);
        a = an;
#pragma unroll
        for (int j = 0; j < 4; ++j) b[j] = bn[j];
    }
#pragma unroll
    for (int j = 0; j < 4; ++j)
        acc[j] = __builtin_amdgcn_wmma_f32_16x16x32_bf16(
            false, a, false, b[j], (short)0, acc[j], false, false);

#pragma unroll
    for (int j = 0; j < 4; ++j) {
        int n = nb + j * 16 + l15;
        float bias = bih[n] + bhh[n];
#pragma unroll
        for (int r = 0; r < 8; ++r) {
            int m = mb + hi * 8 + r;
            xg[(size_t)m * G4 + n] = acc[j][r] + bias;
        }
    }
}

// ----------------------------------------------- persistent LSTM recurrence
// One 1024-thread workgroup.  h (bf16) lives in 64KB LDS, c in registers.
__global__ __launch_bounds__(1024) void lstm_kernel(
    const float* __restrict__ xg, const bf16* __restrict__ whh,
    float* __restrict__ gates, bf16* __restrict__ hs) {
    __shared__ bf16 hsm[BATCH * HIDDEN];   // 64 KB

    int tid = threadIdx.x;
    int lane = tid & 31, wave = tid >> 5;
    int l15 = lane & 15, hi = lane >> 4;

    for (int i = tid; i < BATCH * HIDDEN; i += 1024) hsm[i] = (bf16)0.0f;
    float c[32];
#pragma unroll
    for (int j = 0; j < 32; ++j) c[j] = 0.0f;
    __syncthreads();

    for (int t = 0; t < SEQ; ++t) {
        const float* xgt = xg + (size_t)t * BATCH * G4;
        // ---- phase A: gates = x_gates[t] + h @ W_hh^T  (M=64,N=2048,K=512)
        for (int tile = wave; tile < 512; tile += 32) {
            int mt = tile & 3, nt = tile >> 2;
            int mb = mt * 16, nb = nt * 16;
            v8f acc;
#pragma unroll
            for (int r = 0; r < 8; ++r)
                acc[r] = xgt[(size_t)(mb + hi * 8 + r) * G4 + nb + l15];
            const bf16* arow = hsm + (mb + l15) * HIDDEN;            // LDS
            const bf16* brow = whh + (size_t)(nb + l15) * HIDDEN;    // global
#pragma unroll 4
            for (int k = 0; k < HIDDEN; k += 32) {
                v16bf a = load_frag(arow + k, hi);
                v16bf b = load_frag(brow + k, hi);
                acc = __builtin_amdgcn_wmma_f32_16x16x32_bf16(
                    false, a, false, b, (short)0, acc, false, false);
            }
#pragma unroll
            for (int r = 0; r < 8; ++r)
                gates[(size_t)(mb + hi * 8 + r) * G4 + nb + l15] = acc[r];
        }
        __syncthreads();
        // ---- phase B: activations, state update
#pragma unroll
        for (int j = 0; j < 32; ++j) {
            int e = j * 1024 + tid;          // 0..32767 -> (b, h)
            int b = e >> 9, h = e & 511;
            const float* gb = gates + (size_t)b * G4;
            float gi = gb[h];
            float gf = gb[HIDDEN + h];
            float gg = gb[2 * HIDDEN + h];
            float go = gb[3 * HIDDEN + h];
            float cn = sigmoidf(gf) * c[j] + sigmoidf(gi) * tanhf(gg);
            c[j] = cn;
            float hn = sigmoidf(go) * tanhf(cn);
            bf16 hb = (bf16)hn;
            hsm[b * HIDDEN + h] = hb;
            hs[(size_t)(t * BATCH + b) * HIDDEN + h] = hb;
        }
        __syncthreads();
    }
}

// --------------------------------------------- logits = hs @ W_fc^T + b_fc
// M=2048, N=10000, K=512.  Block = one 16-row M-tile x 512 columns (8 waves
// x 64 cols).  The 16x512 bf16 A-tile (16 KB) is staged into LDS ONCE per
// block by the Tensor Data Mover (tensor_load_to_lds), then every wave reads
// A fragments from LDS while streaming W_fc from global (double-buffered).
__global__ __launch_bounds__(256) void logits_gemm_kernel(
    const bf16* __restrict__ hsb, const bf16* __restrict__ wfc,
    const float* __restrict__ bfc, float* __restrict__ out) {
    __shared__ bf16 atile[16 * HIDDEN];       // 16 KB

    const int NCG = 20;                       // column groups of 8*64 = 512
    int mt = blockIdx.x / NCG;                // 0..127
    int cg = blockIdx.x % NCG;                // 0..19
    int mb = mt * 16;
    int wave = threadIdx.x >> 5;              // 0..7
    int lane = threadIdx.x & 31;
    int l15 = lane & 15, hi = lane >> 4;

    // ---- TDM: stage A-tile (16 rows x 512 cols, bf16) into LDS ----
    if (wave == 0) {
        unsigned long long ga =
            (unsigned long long)(size_t)hsb + (unsigned long long)mb * HIDDEN * 2;
        unsigned lds_base = (unsigned)(size_t)(&atile[0]);  // low 32 = LDS offset
        u32x4 g0;
        g0[0] = 1u;                                    // count=1 valid descriptor
        g0[1] = lds_base;                              // lds_addr
        g0[2] = (unsigned)ga;                          // global_addr[31:0]
        g0[3] = (unsigned)((ga >> 32) & 0x1FFFFFFu)    // global_addr[56:32]
                | (2u << 30);                          // type=2 ("image")
        u32x8 g1;
        g1[0] = 1u << 16;                              // data_size=1 (2 bytes)
        g1[1] = (unsigned)HIDDEN << 16;                // tensor_dim0 lo16 @[63:48]
        g1[2] = (16u << 16);                           // dim0 hi16=0 | tensor_dim1=16 lo16
        g1[3] = ((unsigned)HIDDEN << 16);              // dim1 hi16=0 | tile_dim0=512
        g1[4] = 16u;                                   // tile_dim1=16, tile_dim2=0
        g1[5] = (unsigned)HIDDEN;                      // tensor_dim0_stride lo32
        g1[6] = 0u;                                    // stride hi | dim1_stride lo
        g1[7] = 0u;
        asm volatile("tensor_load_to_lds %0, %1" :: "s"(g0), "s"(g1) : "memory");
        __builtin_amdgcn_s_wait_tensorcnt(0);
    }
    __syncthreads();

    int ng = cg * 8 + wave;                   // 0..159
    int nb0 = ng * 64;
    int ntiles = 0;
    if (nb0 < VOCAB) {
        ntiles = (VOCAB - nb0) / 16;          // all N-tiles are full (625*16)
        if (ntiles > 4) ntiles = 4;
    }
    if (ntiles <= 0) return;                  // wave-uniform

    v8f acc[4] = {};
    const bf16* arow = atile + l15 * HIDDEN;  // LDS
    const bf16* brow[4];
    for (int j = 0; j < ntiles; ++j)
        brow[j] = wfc + (size_t)(nb0 + j * 16 + l15) * HIDDEN;

    v16bf b[4];
    for (int j = 0; j < ntiles; ++j) b[j] = load_frag(brow[j], hi);
    v16bf a = load_frag(arow, hi);

#pragma unroll 2
    for (int k = 32; k < HIDDEN; k += 32) {
        v16bf bn[4];
        for (int j = 0; j < ntiles; ++j) bn[j] = load_frag(brow[j] + k, hi);
        v16bf an = load_frag(arow + k, hi);
        for (int j = 0; j < ntiles; ++j)
            acc[j] = __builtin_amdgcn_wmma_f32_16x16x32_bf16(
                false, a, false, b[j], (short)0, acc[j], false, false);
        a = an;
        for (int j = 0; j < ntiles; ++j) b[j] = bn[j];
    }
    for (int j = 0; j < ntiles; ++j)
        acc[j] = __builtin_amdgcn_wmma_f32_16x16x32_bf16(
            false, a, false, b[j], (short)0, acc[j], false, false);

    for (int j = 0; j < ntiles; ++j) {
        int n = nb0 + j * 16 + l15;
        float bias = bfc[n];
#pragma unroll
        for (int r = 0; r < 8; ++r) {
            int m = mb + hi * 8 + r;          // row = t*64 + b
            int bb = m & 63, tt = m >> 6;
            out[((size_t)bb * SEQ + tt) * VOCAB + n] = acc[j][r] + bias;
        }
    }
}

// ---------------------------------------------------------------------------
extern "C" void kernel_launch(void* const* d_in, const int* in_sizes, int n_in,
                              void* d_out, int out_size, void* d_ws, size_t ws_size,
                              hipStream_t stream) {
    const float* features = (const float*)d_in[0];
    const int*   captions = (const int*)d_in[1];   // JAX default: int32
    const float* emb      = (const float*)d_in[2];
    const float* W_ih     = (const float*)d_in[3];
    const float* W_hh     = (const float*)d_in[4];
    const float* b_ih     = (const float*)d_in[5];
    const float* b_hh     = (const float*)d_in[6];
    const float* W_fc     = (const float*)d_in[7];
    const float* b_fc     = (const float*)d_in[8];
    float* out = (float*)d_out;

    char* ws = (char*)d_ws;
    size_t off = 0;
    auto alloc = [&](size_t bytes) -> void* {
        void* p = ws + off;
        off = (off + bytes + 255) & ~(size_t)255;
        return p;
    };
    bf16*  xs_bf  = (bf16*)alloc((size_t)MROWS * EMBED * 2);     //  2 MB
    bf16*  wih_bf = (bf16*)alloc((size_t)G4 * EMBED * 2);        //  2 MB
    bf16*  whh_bf = (bf16*)alloc((size_t)G4 * HIDDEN * 2);       //  2 MB
    bf16*  wfc_bf = (bf16*)alloc((size_t)VOCAB * HIDDEN * 2);    // 10.2 MB
    float* xg     = (float*)alloc((size_t)MROWS * G4 * 4);       // 16.8 MB
    bf16*  hs_bf  = (bf16*)alloc((size_t)MROWS * HIDDEN * 2);    //  2 MB
    float* gates  = (float*)alloc((size_t)BATCH * G4 * 4);       //  0.5 MB

    // 1) bf16 conversions of the three weight matrices
    cvt_bf16_kernel<<<1024, 256, 0, stream>>>(W_ih, wih_bf, G4 * EMBED);
    cvt_bf16_kernel<<<1024, 256, 0, stream>>>(W_hh, whh_bf, G4 * HIDDEN);
    cvt_bf16_kernel<<<2048, 256, 0, stream>>>(W_fc, wfc_bf, VOCAB * HIDDEN);

    // 2) embedding gather + feature row -> xs (bf16)
    build_xs_kernel<<<1024, 256, 0, stream>>>(features, captions, emb, xs_bf);

    // 3) x_gates GEMM: 4096 waves (128 M-tiles x 32 N-groups), 8 waves/block
    xgates_gemm_kernel<<<512, 256, 0, stream>>>(xs_bf, wih_bf, b_ih, b_hh, xg);

    // 4) persistent LSTM recurrence (single workgroup, h in LDS)
    lstm_kernel<<<1, 1024, 0, stream>>>(xg, whh_bf, gates, hs_bf);

    // 5) logits GEMM: 128 M-tiles x 20 column-groups = 2560 blocks
    logits_gemm_kernel<<<2560, 256, 0, stream>>>(hs_bf, wfc_bf, b_fc, out);
}
